// ContrastiveLoss_77919296684640
// MI455X (gfx1250) — matrix operand order repaired and checked by reading, not compiled, not run on hardware
//
#include <hip/hip_runtime.h>

typedef __attribute__((ext_vector_type(16))) _Float16 v16h;
typedef __attribute__((ext_vector_type(8)))  _Float16 v8h;
typedef __attribute__((ext_vector_type(4)))  _Float16 v4h;
typedef __attribute__((ext_vector_type(8)))  float    v8f;
typedef __attribute__((ext_vector_type(4)))  int      v4i;

#define BN        8192
#define DIM       128
#define NROWS     16384                      // 2B
#define INV_T     14.285714285714286f        // 1/0.07, fixed LSE max bound
#define C_EXP2    20.60992915555662f         // (1/T) * log2(e)
#define CSPLITS   8
#define COLS_PER_SPLIT (NROWS / CSPLITS)     // 2048
#define STAGE_COLS 64
#define NSTAGES   (COLS_PER_SPLIT / STAGE_COLS)  // 32
#define LDS_STRIDE 136                       // 128 halfs + 8 pad -> conflict-free

#define AS1G __attribute__((address_space(1)))
#define AS3L __attribute__((address_space(3)))

#if defined(__AMDGCN__) && \
    __has_builtin(__builtin_amdgcn_global_load_async_to_lds_b128) && \
    __has_builtin(__builtin_amdgcn_s_wait_asynccnt)
#define USE_ASYNC_LDS 1
// global side: generic and AS1 share address values on amdgcn; integer
// laundering sidesteps const/addrspace conversion rules.
#define GPTR(p) ((AS1G v4i*)(unsigned long long)(p))
// LDS side: needs a real addrspacecast (generic LDS ptr carries the aperture).
#define LPTR(p) ((AS3L v4i*)(p))
#else
#define USE_ASYNC_LDS 0
#endif

#if defined(__AMDGCN__) && __has_builtin(__builtin_amdgcn_exp2f)
#define EXP2F(x) __builtin_amdgcn_exp2f(x)
#else
#define EXP2F(x) exp2f(x)
#endif

// ---------------------------------------------------------------------------
// Kernel 1: L2-normalize, emit f16 rows E[0..B-1]=a_hat, E[B..2B-1]=b_hat,
// and exact-fp32 positive logits pos[i] = <a,b>/(|a||b| T). One wave per pair.
// ---------------------------------------------------------------------------
__global__ __launch_bounds__(256) void nrm_kernel(const float* __restrict__ a,
                                                  const float* __restrict__ b,
                                                  _Float16* __restrict__ E,
                                                  float* __restrict__ pos)
{
    const int lane = threadIdx.x & 31;
    const int wave = threadIdx.x >> 5;
    const int i = blockIdx.x * 8 + wave;

    const float4 av = ((const float4*)a)[i * 32 + lane];
    const float4 bv = ((const float4*)b)[i * 32 + lane];

    float sa = av.x*av.x + av.y*av.y + av.z*av.z + av.w*av.w;
    float sb = bv.x*bv.x + bv.y*bv.y + bv.z*bv.z + bv.w*bv.w;
    float dp = av.x*bv.x + av.y*bv.y + av.z*bv.z + av.w*bv.w;
    #pragma unroll
    for (int off = 16; off >= 1; off >>= 1) {
        sa += __shfl_xor(sa, off, 32);
        sb += __shfl_xor(sb, off, 32);
        dp += __shfl_xor(dp, off, 32);
    }
    const float na = fmaxf(sqrtf(sa), 1e-12f);
    const float nb = fmaxf(sqrtf(sb), 1e-12f);
    const float ia = 1.0f / na;
    const float ib = 1.0f / nb;

    v4h ea = { (_Float16)(av.x*ia), (_Float16)(av.y*ia),
               (_Float16)(av.z*ia), (_Float16)(av.w*ia) };
    v4h eb = { (_Float16)(bv.x*ib), (_Float16)(bv.y*ib),
               (_Float16)(bv.z*ib), (_Float16)(bv.w*ib) };
    ((v4h*)(E + (size_t)i * DIM))[lane]        = ea;
    ((v4h*)(E + (size_t)(i + BN) * DIM))[lane] = eb;
    if (lane == 0) pos[i] = dp * ia * ib * INV_T;
}

// ---------------------------------------------------------------------------
// Kernel 2: fused sim + sum(exp(logit - 1/T)), no sim materialization.
// grid = (128 row-blocks, 8 column-splits), 256 threads = 8 waves.
// Double-buffered async global->LDS staging; f16 WMMA 16x16x32, K=128.
// Diagonal (self-similarity) is NOT masked in the hot loop; its contribution
// is subtracted afterwards from the per-slot sums (branch-free inner loop).
// ---------------------------------------------------------------------------
__global__ __launch_bounds__(256) void lse_kernel(const _Float16* __restrict__ E,
                                                  float* __restrict__ partial)
{
    __shared__ __align__(16) _Float16 lds[2][STAGE_COLS * LDS_STRIDE];

    const int tid  = threadIdx.x;
    const int lane = tid & 31;
    // force wave id into an SGPR so Rbase math / branches are scalar
    const int wave = __builtin_amdgcn_readfirstlane(tid >> 5);
    const int l15  = lane & 15;
    const int hsel = lane >> 4;
    const int Rbase = blockIdx.x * 128 + wave * 16;
    const int split = blockIdx.y;
    const int Cbeg  = split * COLS_PER_SPLIT;

    // This thread's 4 staging chunks (8 halfs each): fixed LDS offsets,
    // global pointers advance by STAGE_COLS*DIM per stage.
    unsigned ldoff[4];
    const _Float16* gptr[4];
    #pragma unroll
    for (int j = 0; j < 4; ++j) {
        const int chunk = tid + 256 * j;
        const int r  = chunk >> 4;
        const int c8 = chunk & 15;
        ldoff[j] = (unsigned)(r * LDS_STRIDE + c8 * 8);
        gptr[j]  = E + (size_t)(Cbeg + r) * DIM + c8 * 8;
    }

    // A operand: 16-bit A 16x32 in contiguous k order, same as B.
    // lanes 0-15: M=l, K=0..15; lanes 16-31: M=l-16, K=16..31
    v16h A[4];
    {
        const _Float16* arow = E + (size_t)(Rbase + l15) * DIM + hsel * 16;
        #pragma unroll
        for (int kk = 0; kk < 4; ++kk) {
            v8h lo = *(const v8h*)(arow + kk * 32);
            v8h hi = *(const v8h*)(arow + kk * 32 + 8);
            #pragma unroll
            for (int j = 0; j < 8; ++j) { A[kk][j] = lo[j]; A[kk][8 + j] = hi[j]; }
        }
    }

    float s[8];
    #pragma unroll
    for (int p = 0; p < 8; ++p) s[p] = 0.0f;

#if USE_ASYNC_LDS
    // prefetch stage 0 into buffer 0
    #pragma unroll
    for (int j = 0; j < 4; ++j)
        __builtin_amdgcn_global_load_async_to_lds_b128(
            GPTR(gptr[j]), LPTR(&lds[0][ldoff[j]]), 0, 0);
#endif

    for (int st = 0; st < NSTAGES; ++st) {
        const int cur = st & 1;
#if USE_ASYNC_LDS
        if (st + 1 < NSTAGES) {
            #pragma unroll
            for (int j = 0; j < 4; ++j)
                __builtin_amdgcn_global_load_async_to_lds_b128(
                    GPTR(gptr[j] + (size_t)(st + 1) * STAGE_COLS * DIM),
                    LPTR(&lds[cur ^ 1][ldoff[j]]), 0, 0);
            __builtin_amdgcn_s_wait_asynccnt(4);   // stage st landed; st+1 in flight
        } else {
            __builtin_amdgcn_s_wait_asynccnt(0);
        }
        __syncthreads();
#else
        __syncthreads();
        {
            v8h t0 = *(const v8h*)(gptr[0] + (size_t)st * STAGE_COLS * DIM);
            v8h t1 = *(const v8h*)(gptr[1] + (size_t)st * STAGE_COLS * DIM);
            v8h t2 = *(const v8h*)(gptr[2] + (size_t)st * STAGE_COLS * DIM);
            v8h t3 = *(const v8h*)(gptr[3] + (size_t)st * STAGE_COLS * DIM);
            *(v8h*)&lds[cur][ldoff[0]] = t0;
            *(v8h*)&lds[cur][ldoff[1]] = t1;
            *(v8h*)&lds[cur][ldoff[2]] = t2;
            *(v8h*)&lds[cur][ldoff[3]] = t3;
        }
        __syncthreads();
#endif

        #pragma unroll
        for (int ct = 0; ct < 4; ++ct) {
            // B operand (32x16): lane n<16 -> col n, K=0..15; lane 16+n -> col n, K=16..31
            const _Float16* brow = &lds[cur][(ct * 16 + l15) * LDS_STRIDE + hsel * 16];
            v16h B0 = *(const v16h*)(brow);
            v16h B1 = *(const v16h*)(brow + 32);
            v16h B2 = *(const v16h*)(brow + 64);
            v16h B3 = *(const v16h*)(brow + 96);
            v8f acc0 = {0.f,0.f,0.f,0.f,0.f,0.f,0.f,0.f};
            v8f acc1 = {0.f,0.f,0.f,0.f,0.f,0.f,0.f,0.f};
            acc0 = __builtin_amdgcn_wmma_f32_16x16x32_f16(false, A[0], false, B0, (short)0, acc0, false, false);
            acc1 = __builtin_amdgcn_wmma_f32_16x16x32_f16(false, A[1], false, B1, (short)0, acc1, false, false);
            acc0 = __builtin_amdgcn_wmma_f32_16x16x32_f16(false, A[2], false, B2, (short)0, acc0, false, false);
            acc1 = __builtin_amdgcn_wmma_f32_16x16x32_f16(false, A[3], false, B3, (short)0, acc1, false, false);

            // C/D layout: lane l slot p holds (M = p + 8*hsel, N = l15)
            #pragma unroll
            for (int p = 0; p < 8; ++p)
                s[p] += EXP2F((acc0[p] + acc1[p]) * C_EXP2 - C_EXP2);
        }

        __syncthreads();   // protect lds[cur] before it is refilled at st+2
    }

    // Subtract the (un-masked) diagonal contribution if this split's column
    // range contains this wave's rows. d_m = sum_k E[m,k]^2 over the same f16
    // values WMMA consumed; lane + its partner (lane^16) hold row l15's K's.
    if (Rbase >= Cbeg && Rbase < Cbeg + COLS_PER_SPLIT) {   // scalar branch
        float ssql = 0.0f;
        #pragma unroll
        for (int kk = 0; kk < 4; ++kk)
            #pragma unroll
            for (int j = 0; j < 16; ++j) {
                const float x = (float)A[kk][j];
                ssql += x * x;
            }
        const float ssq  = ssql + __shfl_xor(ssql, 16, 32);  // full row-l15 norm^2
        const float corr = EXP2F((ssq - 1.0f) * C_EXP2);
        #pragma unroll
        for (int p = 0; p < 8; ++p)
            if (l15 == p + 8 * hsel) s[p] -= corr;   // the one slot where row==col
    }

    // row sums live across the 16 lanes of each half-wave
    #pragma unroll
    for (int p = 0; p < 8; ++p) {
        float r = s[p];
        r += __shfl_xor(r, 1, 32);
        r += __shfl_xor(r, 2, 32);
        r += __shfl_xor(r, 4, 32);
        r += __shfl_xor(r, 8, 32);
        if (l15 == 0) {
            const int grow = Rbase + hsel * 8 + p;
            partial[(size_t)split * NROWS + grow] = r;
        }
    }
}

// ---------------------------------------------------------------------------
// Kernel 3: deterministic finalize. Per row: sum 8 split-partials,
// loss_i = (1/T + log(sum)) - pos, fixed-order tree mean.
// ---------------------------------------------------------------------------
__global__ __launch_bounds__(256) void fin_kernel(const float* __restrict__ partial,
                                                  const float* __restrict__ pos,
                                                  float* __restrict__ out)
{
    __shared__ float sm[256];
    const int tid = threadIdx.x;
    float acc = 0.0f;
    for (int j = 0; j < NROWS / 256; ++j) {
        const int row = tid + 256 * j;
        float ssum = 0.0f;
        #pragma unroll
        for (int k = 0; k < CSPLITS; ++k) ssum += partial[(size_t)k * NROWS + row];
        const int pi = (row < BN) ? row : row - BN;
        acc += (INV_T + __logf(ssum)) - pos[pi];
    }
    sm[tid] = acc;
    __syncthreads();
    for (int w = 128; w > 0; w >>= 1) {
        if (tid < w) sm[tid] += sm[tid + w];
        __syncthreads();
    }
    if (tid == 0) out[0] = sm[0] * (1.0f / NROWS);
}

extern "C" void kernel_launch(void* const* d_in, const int* in_sizes, int n_in,
                              void* d_out, int out_size, void* d_ws, size_t ws_size,
                              hipStream_t stream) {
    (void)in_sizes; (void)n_in; (void)out_size; (void)ws_size;
    const float* a = (const float*)d_in[0];
    const float* b = (const float*)d_in[1];
    float* out = (float*)d_out;

    char* ws = (char*)d_ws;
    _Float16* E  = (_Float16*)ws;                                       // 4 MB
    float* pos   = (float*)(ws + (size_t)NROWS * DIM * 2);              // 32 KB
    float* part  = (float*)(ws + (size_t)NROWS * DIM * 2 + BN * 4);     // 512 KB

    nrm_kernel<<<dim3(BN / 8), dim3(256), 0, stream>>>(a, b, E, pos);
    lse_kernel<<<dim3(NROWS / 128, CSPLITS), dim3(256), 0, stream>>>(E, part);
    fin_kernel<<<dim3(1), dim3(256), 0, stream>>>(part, pos, out);
}
